// GeoSemNodeEm_64252710748377
// MI455X (gfx1250) — matrix operand order repaired and checked by reading, not compiled, not run on hardware
//
#include <hip/hip_runtime.h>

// Problem constants
#define NN 1024
#define TT 12
#define CC 32
#define DD 64
#define HH 256
#define OO 64
#define MROWS (NN * TT * CC)      // 393216
#define ROWS_PER_BLOCK 512        // 8 waves x 4 tiles x 16 rows
#define TILES_PER_WAVE 4

typedef __attribute__((ext_vector_type(16))) __bf16 v16bf;
typedef __attribute__((ext_vector_type(8)))  __bf16 v8bf;
typedef __attribute__((ext_vector_type(8)))  float  v8f;
typedef __attribute__((ext_vector_type(4)))  float  vf4;

__device__ __forceinline__ float relu1(float v) {
    return __builtin_amdgcn_fmed3f(v, 0.0f, __builtin_inff());   // 1 VALU relu
}

// ---------------------------------------------------------------------------
// Kernel 1: colsum[t][j] = sum_i A[t][i][j]   (A = semantic_data[:,0])
// ---------------------------------------------------------------------------
__global__ void colsum_kernel(const float* __restrict__ A, float* __restrict__ colsums) {
    int idx = blockIdx.x * blockDim.x + threadIdx.x;   // t*N + j
    if (idx >= TT * NN) return;
    int t = idx >> 10;
    int j = idx & (NN - 1);
    const float* p = A + (size_t)t * NN * NN + j;
    float s = 0.0f;
#pragma unroll 8
    for (int i = 0; i < NN; ++i) s += p[(size_t)i * NN];
    colsums[idx] = s;
}

// ---------------------------------------------------------------------------
// Kernel 2: fused geo-scale + 3-layer MLP, bf16 WMMA, f32 accumulate.
// Transposed orientation: D = W^T (A-op) x Act^T (B-op), bias as C-init.
// Each wave: 4 tiles of 16 rows (rolled loop); 8 waves/block; grid = 768.
// ---------------------------------------------------------------------------
__launch_bounds__(256)
__global__ void fused_mlp_kernel(const float* __restrict__ x,
                                 const float* __restrict__ colsums,
                                 const float* __restrict__ W1, const float* __restrict__ b1,
                                 const float* __restrict__ W2, const float* __restrict__ b2,
                                 const float* __restrict__ Wo, const float* __restrict__ bo,
                                 float* __restrict__ out) {
    // Transposed bf16 weights in LDS: WT[feat][k], K contiguous.
    __shared__ __bf16 w1t[HH * DD];    // [h][d]   32 KB
    __shared__ __bf16 w2t[OO * HH];    // [o][h]   32 KB
    __shared__ __bf16 wot[OO * OO];    // [o][k]    8 KB
    __shared__ __attribute__((aligned(32))) float b1s[HH];
    __shared__ __attribute__((aligned(32))) float b2s[OO];
    __shared__ __attribute__((aligned(32))) float bos[OO];
    // Per-wave staging (h1 16x256 bf16; h2 overlaps, guarded by reg hoisting)
    __shared__ __attribute__((aligned(32))) char stage[8 * 8192];   // 64 KB

    const int tid    = threadIdx.x;
    const int wv     = tid >> 5;
    const int lane   = tid & 31;
    const int lane16 = lane & 15;
    const int lhalf  = lane >> 4;

    // ---- cooperative weight load (float4) + transpose + bf16 convert ----
    {
        const vf4* W14 = (const vf4*)W1;
#pragma unroll
        for (int it = 0; it < (DD * HH / 4) / 256; ++it) {   // 16 iters
            int i4 = it * 256 + tid;
            vf4 v = W14[i4];
            int d = (i4 * 4) >> 8, h = (i4 * 4) & (HH - 1);
#pragma unroll
            for (int c = 0; c < 4; ++c) w1t[(h + c) * DD + d] = (__bf16)v[c];
        }
        const vf4* W24 = (const vf4*)W2;
#pragma unroll
        for (int it = 0; it < (HH * OO / 4) / 256; ++it) {   // 16 iters
            int i4 = it * 256 + tid;
            vf4 v = W24[i4];
            int h = (i4 * 4) >> 6, o = (i4 * 4) & (OO - 1);
#pragma unroll
            for (int c = 0; c < 4; ++c) w2t[(o + c) * HH + h] = (__bf16)v[c];
        }
        const vf4* Wo4 = (const vf4*)Wo;
#pragma unroll
        for (int it = 0; it < (OO * OO / 4) / 256; ++it) {   // 4 iters
            int i4 = it * 256 + tid;
            vf4 v = Wo4[i4];
            int k = (i4 * 4) >> 6, o = (i4 * 4) & (OO - 1);
#pragma unroll
            for (int c = 0; c < 4; ++c) wot[(o + c) * OO + k] = (__bf16)v[c];
        }
    }
    if (tid < HH) b1s[tid] = b1[tid];
    if (tid < OO) { b2s[tid] = b2[tid]; bos[tid] = bo[tid]; }
    __syncthreads();

    __bf16* sH = (__bf16*)(stage + wv * 8192);   // h1 [16][256]; h2 overlaps

#pragma unroll 1
    for (int tile = 0; tile < TILES_PER_WAVE; ++tile) {
        const int r0 = blockIdx.x * ROWS_PER_BLOCK + wv * 64 + tile * 16;
        const float denom  = colsums[((r0 / CC) % TT) * NN + r0 / (TT * CC)];
        const float factor = (denom != 0.0f) ? 2.0f : 0.0f;  // x + (denom*x)/denom

        // ---- B1 fragments direct from global x (scale + bf16 in regs) ----
        v16bf bG[2];
        {
            const vf4* xrow = (const vf4*)(x + (size_t)(r0 + lane16) * DD) + lhalf * 4;
#pragma unroll
            for (int ks = 0; ks < 2; ++ks) {
                v16bf g;
#pragma unroll
                for (int q = 0; q < 4; ++q) {
                    vf4 v = xrow[ks * 8 + q];
#pragma unroll
                    for (int c = 0; c < 4; ++c) g[q * 4 + c] = (__bf16)(v[c] * factor);
                }
                bG[ks] = g;
            }
        }

        // ---- GEMM1: h1^T = relu(W1^T @ geo^T + b1) -> LDS [row][h] ----
        {
            const __bf16* pA = w1t + lane16 * DD + lhalf * 16;
            __bf16*       pS = sH + lane16 * HH + lhalf * 8;
            const float*  pb = b1s + lhalf * 8;
#pragma unroll
            for (int mt = 0; mt < 16; ++mt) {
                v8f acc = *(const v8f*)(pb + mt * 16);           // bias init
#pragma unroll
                for (int ks = 0; ks < 2; ++ks) {
                    v8bf lo = *(const v8bf*)(pA + mt * 16 * DD + ks * 32);
                    v8bf hi = *(const v8bf*)(pA + mt * 16 * DD + ks * 32 + 8);
                    v16bf a = __builtin_shufflevector(lo, hi, 0,1,2,3,4,5,6,7,8,9,10,11,12,13,14,15);
                    acc = __builtin_amdgcn_wmma_f32_16x16x32_bf16(false, a, false, bG[ks],
                                                                  (short)0, acc, false, false);
                }
                v8bf hb;
#pragma unroll
                for (int v = 0; v < 8; ++v) hb[v] = (__bf16)relu1(acc[v]);
                *(v8bf*)(pS + mt * 16) = hb;                     // one b128 store
            }
        }
        asm volatile("" ::: "memory");

        // ---- B2 fragments: h1^T, eight contiguous 32B reads ----
        v16bf bH[8];
        {
            const __bf16* pB = sH + lane16 * HH + lhalf * 16;
#pragma unroll
            for (int ks = 0; ks < 8; ++ks)
                bH[ks] = *(const v16bf*)(pB + ks * 32);
        }
        asm volatile("" ::: "memory");

        // ---- GEMM2: h2^T = relu(W2^T @ h1^T + b2) -> LDS [row][o] ----
        {
            const __bf16* pA = w2t + lane16 * HH + lhalf * 16;
            __bf16*       pS = sH + lane16 * OO + lhalf * 8;     // h2 (overlaps h1)
            const float*  pb = b2s + lhalf * 8;
#pragma unroll
            for (int mt = 0; mt < 4; ++mt) {
                v8f acc = *(const v8f*)(pb + mt * 16);           // bias init
#pragma unroll
                for (int ks = 0; ks < 8; ++ks) {
                    v8bf lo = *(const v8bf*)(pA + mt * 16 * HH + ks * 32);
                    v8bf hi = *(const v8bf*)(pA + mt * 16 * HH + ks * 32 + 8);
                    v16bf a = __builtin_shufflevector(lo, hi, 0,1,2,3,4,5,6,7,8,9,10,11,12,13,14,15);
                    acc = __builtin_amdgcn_wmma_f32_16x16x32_bf16(false, a, false, bH[ks],
                                                                  (short)0, acc, false, false);
                }
                v8bf hb;
#pragma unroll
                for (int v = 0; v < 8; ++v) hb[v] = (__bf16)relu1(acc[v]);
                *(v8bf*)(pS + mt * 16) = hb;
            }
        }
        asm volatile("" ::: "memory");

        // ---- B3 fragments: h2^T ----
        v16bf bH2[2];
        {
            const __bf16* pB = sH + lane16 * OO + lhalf * 16;
#pragma unroll
            for (int ks = 0; ks < 2; ++ks)
                bH2[ks] = *(const v16bf*)(pB + ks * 32);
        }
        asm volatile("" ::: "memory");

        // ---- GEMM3: out^T = Wo^T @ h2^T + bo, direct v8f global stores ----
        {
            const __bf16* pA = wot + lane16 * OO + lhalf * 16;
            const float*  pb = bos + lhalf * 8;
            float* orow = out + (size_t)(r0 + lane16) * OO + lhalf * 8;
#pragma unroll
            for (int mt = 0; mt < 4; ++mt) {
                v8f acc = *(const v8f*)(pb + mt * 16);           // bias init
#pragma unroll
                for (int ks = 0; ks < 2; ++ks) {
                    v8bf lo = *(const v8bf*)(pA + mt * 16 * OO + ks * 32);
                    v8bf hi = *(const v8bf*)(pA + mt * 16 * OO + ks * 32 + 8);
                    v16bf a = __builtin_shufflevector(lo, hi, 0,1,2,3,4,5,6,7,8,9,10,11,12,13,14,15);
                    acc = __builtin_amdgcn_wmma_f32_16x16x32_bf16(false, a, false, bH2[ks],
                                                                  (short)0, acc, false, false);
                }
                *(v8f*)(orow + mt * 16) = acc;                   // 32B contiguous/lane
            }
        }
        asm volatile("" ::: "memory");   // order LDS reuse across tile iterations
    }
}

// ---------------------------------------------------------------------------
extern "C" void kernel_launch(void* const* d_in, const int* in_sizes, int n_in,
                              void* d_out, int out_size, void* d_ws, size_t ws_size,
                              hipStream_t stream) {
    const float* x   = (const float*)d_in[0];
    const float* sem = (const float*)d_in[1];
    const float* W1  = (const float*)d_in[2];
    const float* b1  = (const float*)d_in[3];
    const float* W2  = (const float*)d_in[4];
    const float* b2  = (const float*)d_in[5];
    const float* Wo  = (const float*)d_in[6];
    const float* bo  = (const float*)d_in[7];
    float* out     = (float*)d_out;
    float* colsums = (float*)d_ws;   // T*N floats = 48 KB

    colsum_kernel<<<(TT * NN + 255) / 256, 256, 0, stream>>>(sem, colsums);
    fused_mlp_kernel<<<MROWS / ROWS_PER_BLOCK, 256, 0, stream>>>(x, colsums,
                                                                 W1, b1, W2, b2, Wo, bo, out);
}
